// ContrastiveLoss_80461917323497
// MI455X (gfx1250) — matrix operand order repaired and checked by reading, not compiled, not run on hardware
//
#include <hip/hip_runtime.h>

#define NC    8
#define NOBJ  4096
#define DIM   512
#define NLVL  3
#define MARGINF 1.3f
#define TEMPF   0.01f
#define EPSF    1e-8f

typedef __attribute__((ext_vector_type(16))) __bf16 v16bf;
typedef __attribute__((ext_vector_type(8)))  float  v8f;

// ---- workspace layout (bytes) ----
#define WS_COUNTS 0                         // 8 * u32
#define WS_WITHIN 32                        // 3*8 f32
#define WS_CESUM  128                       // 3*8 f32
#define WS_RNORM  256                       // 3*4096 f32
#define WS_MEANS  (256 + NLVL*NOBJ*4)       // 3*8*512 f32
#define WS_BF16   (WS_MEANS + NLVL*NC*DIM*4)        // 3*4096*512 bf16 (normalized rows)
#define WS_TOTAL  (WS_BF16 + (size_t)NLVL*NOBJ*DIM*2)

__device__ __forceinline__ unsigned short f2bf(float x) {
    unsigned u = __float_as_uint(x);
    unsigned r = u + 0x7FFFu + ((u >> 16) & 1u);   // round-to-nearest-even
    return (unsigned short)(r >> 16);
}

// ---------------- K0: zero accumulators ----------------
__global__ void k_init(unsigned* counts, float* within, float* cesum) {
    int t = threadIdx.x;
    if (t < NC) counts[t] = 0u;
    if (t < NLVL * NC) { within[t] = 0.f; cesum[t] = 0.f; }
}

// ---------------- K1: row inverse-norms, counts, normalized bf16 mirror ----------------
__global__ void k_prep(const float* __restrict__ f, const int* __restrict__ gt,
                       float* __restrict__ rnorm, unsigned* __restrict__ counts,
                       unsigned short* __restrict__ fb16) {
    int wid  = (blockIdx.x * blockDim.x + threadIdx.x) >> 5;  // one wave per row
    int lane = threadIdx.x & 31;
    if (wid >= NLVL * NOBJ) return;
    const float* row = f + (size_t)wid * DIM;
    float s = 0.f;
    for (int d = lane; d < DIM; d += 32) { float v = row[d]; s += v * v; }
#pragma unroll
    for (int off = 16; off > 0; off >>= 1) s += __shfl_xor(s, off, 32);
    float rn = (s > 0.f) ? rsqrtf(s) : 0.f;
    if (lane == 0) {
        rnorm[wid] = rn;
        if (wid < NOBJ) atomicAdd(&counts[gt[wid]], 1u);   // level 0 only (gt shared)
    }
    if (fb16) {   // normalized bf16 rows for the async WMMA gram kernel
        unsigned short* ob = fb16 + (size_t)wid * DIM;
        for (int d0 = lane * 2; d0 < DIM; d0 += 64) {
            unsigned p = (unsigned)f2bf(row[d0] * rn) |
                         ((unsigned)f2bf(row[d0 + 1] * rn) << 16);
            *(unsigned*)(ob + d0) = p;
        }
    }
}

// ---------------- gram tiling parameters ----------------
#define BM  64
#define BN  64
#define KC  128
#define LDT 136   // padded LDS row length in bf16 elems (16B skew per row)

// ---------------- K2a: Gram cos^2, async global->LDS + bf16 WMMA (fast path) ----------
__device__ __forceinline__ void async_cp16(unsigned ldsOff, unsigned gOff,
                                           unsigned long long base) {
    asm volatile("global_load_async_to_lds_b128 %0, %1, %2 offset:0"
                 :: "v"(ldsOff), "v"(gOff), "s"(base) : "memory");
}

__global__ __launch_bounds__(128) void k_gram_async(const unsigned short* __restrict__ fb16,
                                                    const int* __restrict__ gt,
                                                    float* __restrict__ within) {
    __shared__ unsigned short As[2][BM * LDT];
    __shared__ unsigned short Bs[2][BN * LDT];
    __shared__ float red[NC];

    int bx    = blockIdx.x;
    int l     = bx >> 12;            // 4096 blocks per level
    int rem   = bx & 4095;
    int Mbase = (rem >> 6) << 6;
    int Nbase = (rem & 63) << 6;
    unsigned long long base = (unsigned long long)(size_t)(fb16 + (size_t)l * NOBJ * DIM);

    int tid  = threadIdx.x;
    int lane = tid & 31;
    int w    = tid >> 5;             // wave 0..3 -> 16-row strip of A
    int h    = lane >> 4;            // lane half
    int ln   = lane & 15;

    // per-thread piece of the staged copy: 16B chunk, 8 rows apart per issue
    unsigned chunk = (unsigned)(tid & 15) * 16u;   // byte within 256B row chunk
    int      rbase = tid >> 4;                     // 0..7
    unsigned ldsA0 = (unsigned)(size_t)&As[0][0];
    unsigned ldsA1 = (unsigned)(size_t)&As[1][0];
    unsigned ldsB0 = (unsigned)(size_t)&Bs[0][0];
    unsigned ldsB1 = (unsigned)(size_t)&Bs[1][0];

#define STAGE(ks, ldsA, ldsB)                                                         \
    {                                                                                 \
        _Pragma("unroll") for (int j = 0; j < 8; ++j) {                               \
            int row = j * 8 + rbase;                                                  \
            async_cp16((ldsA) + (unsigned)row * (LDT * 2) + chunk,                    \
                       (unsigned)((Mbase + row) * (DIM * 2) + (ks) * 256) + chunk,    \
                       base);                                                         \
        }                                                                             \
        _Pragma("unroll") for (int j = 0; j < 8; ++j) {                               \
            int row = j * 8 + rbase;                                                  \
            async_cp16((ldsB) + (unsigned)row * (LDT * 2) + chunk,                    \
                       (unsigned)((Nbase + row) * (DIM * 2) + (ks) * 256) + chunk,    \
                       base);                                                         \
        }                                                                             \
    }

    v8f acc[4] = {v8f{}, v8f{}, v8f{}, v8f{}};

    STAGE(0, ldsA0, ldsB0);          // prologue: two stages in flight
    STAGE(1, ldsA1, ldsB1);

#pragma unroll
    for (int ks = 0; ks < DIM / KC; ++ks) {        // 4 stages
        if (ks < 3) asm volatile("s_wait_asynccnt 0x10" ::: "memory");  // stage ks landed
        else        asm volatile("s_wait_asynccnt 0x0"  ::: "memory");
        __syncthreads();

        const unsigned* Au = (const unsigned*)&As[ks & 1][0];
        const unsigned* Bu = (const unsigned*)&Bs[ks & 1][0];
#pragma unroll
        for (int kb = 0; kb < KC; kb += 32) {
            // A fragment: 16x32 bf16 (ISA 16-bit A layout)
            union { unsigned u[8]; v16bf v; } a;
            int abase = ((w * 16 + ln) * LDT + kb) >> 1;
#pragma unroll
            for (int vv = 0; vv < 4; ++vv) a.u[vv] = Au[abase + 4 * h + vv];
#pragma unroll
            for (int vv = 4; vv < 8; ++vv) a.u[vv] = Au[abase + 4 + 4 * h + vv];
#pragma unroll
            for (int t = 0; t < 4; ++t) {
                // B fragment: 32x16 bf16 (same k order as A: K 8h..8h+7, 16+8h..16+8h+7)
                union { unsigned u[8]; v16bf v; } b;
                int bbase = ((t * 16 + ln) * LDT + kb) >> 1;
#pragma unroll
                for (int vv = 0; vv < 8; ++vv) b.u[vv] = Bu[bbase + 4 * h + vv + (vv >= 4 ? 4 : 0)];
                acc[t] = __builtin_amdgcn_wmma_f32_16x16x32_bf16(
                    false, a.v, false, b.v, (short)0, acc[t], false, false);
            }
        }
        __syncthreads();
        if (ks == 0) STAGE(2, ldsA0, ldsB0);       // refill the buffer just consumed
        if (ks == 1) STAGE(3, ldsA1, ldsB1);
    }
#undef STAGE

    // bucket cos^2 by class: only pairs with equal labels contribute
    if (tid < NC) red[tid] = 0.f;
    __syncthreads();
    int cm[8];
#pragma unroll
    for (int r = 0; r < 8; ++r) cm[r] = gt[Mbase + w * 16 + 8 * h + r];
#pragma unroll
    for (int t = 0; t < 4; ++t) {
        int cn = gt[Nbase + t * 16 + ln];
#pragma unroll
        for (int r = 0; r < 8; ++r) {
            float s = acc[t][r];
            if (cm[r] == cn) atomicAdd(&red[cn], s * s);   // ds_add_f32
        }
    }
    __syncthreads();
    if (tid < NC) atomicAdd(&within[l * NC + tid], red[tid]);
}

// ---------------- K2b: fallback (convert-in-kernel) if workspace too small --------
__global__ __launch_bounds__(128) void k_gram_fb(const float* __restrict__ f,
                                                 const int* __restrict__ gt,
                                                 const float* __restrict__ rnorm,
                                                 float* __restrict__ within) {
    __shared__ unsigned short As[BM * LDT];
    __shared__ unsigned short Bs[BN * LDT];
    __shared__ float red[NC];

    int bx    = blockIdx.x;
    int l     = bx >> 12;
    int rem   = bx & 4095;
    int Mbase = (rem >> 6) << 6;
    int Nbase = (rem & 63) << 6;
    const float* fl = f + (size_t)l * NOBJ * DIM;
    const float* rl = rnorm + l * NOBJ;

    int tid  = threadIdx.x;
    int lane = tid & 31;
    int w    = tid >> 5;
    int h    = lane >> 4;
    int ln   = lane & 15;

    v8f acc[4] = {v8f{}, v8f{}, v8f{}, v8f{}};

    for (int ks = 0; ks < DIM; ks += KC) {
        for (int c = tid; c < BM * (KC / 4); c += 128) {
            int r  = c >> 5;
            int kc = (c & 31) << 2;
            {
                int grow = Mbase + r;
                const float4 v = *(const float4*)(fl + (size_t)grow * DIM + ks + kc);
                float rn = rl[grow];
                uint2 p;
                p.x = (unsigned)f2bf(v.x * rn) | ((unsigned)f2bf(v.y * rn) << 16);
                p.y = (unsigned)f2bf(v.z * rn) | ((unsigned)f2bf(v.w * rn) << 16);
                *(uint2*)&As[r * LDT + kc] = p;
            }
            {
                int grow = Nbase + r;
                const float4 v = *(const float4*)(fl + (size_t)grow * DIM + ks + kc);
                float rn = rl[grow];
                uint2 p;
                p.x = (unsigned)f2bf(v.x * rn) | ((unsigned)f2bf(v.y * rn) << 16);
                p.y = (unsigned)f2bf(v.z * rn) | ((unsigned)f2bf(v.w * rn) << 16);
                *(uint2*)&Bs[r * LDT + kc] = p;
            }
        }
        __syncthreads();
        const unsigned* Au = (const unsigned*)As;
        const unsigned* Bu = (const unsigned*)Bs;
#pragma unroll
        for (int kb = 0; kb < KC; kb += 32) {
            union { unsigned u[8]; v16bf v; } a;
            int abase = ((w * 16 + ln) * LDT + kb) >> 1;
#pragma unroll
            for (int vv = 0; vv < 4; ++vv) a.u[vv] = Au[abase + 4 * h + vv];
#pragma unroll
            for (int vv = 4; vv < 8; ++vv) a.u[vv] = Au[abase + 4 + 4 * h + vv];
#pragma unroll
            for (int t = 0; t < 4; ++t) {
                union { unsigned u[8]; v16bf v; } b;
                int bbase = ((t * 16 + ln) * LDT + kb) >> 1;
#pragma unroll
                for (int vv = 0; vv < 8; ++vv) b.u[vv] = Bu[bbase + 4 * h + vv + (vv >= 4 ? 4 : 0)];
                acc[t] = __builtin_amdgcn_wmma_f32_16x16x32_bf16(
                    false, a.v, false, b.v, (short)0, acc[t], false, false);
            }
        }
        __syncthreads();
    }

    if (tid < NC) red[tid] = 0.f;
    __syncthreads();
    int cm[8];
#pragma unroll
    for (int r = 0; r < 8; ++r) cm[r] = gt[Mbase + w * 16 + 8 * h + r];
#pragma unroll
    for (int t = 0; t < 4; ++t) {
        int cn = gt[Nbase + t * 16 + ln];
#pragma unroll
        for (int r = 0; r < 8; ++r) {
            float s = acc[t][r];
            if (cm[r] == cn) atomicAdd(&red[cn], s * s);
        }
    }
    __syncthreads();
    if (tid < NC) atomicAdd(&within[l * NC + tid], red[tid]);
}

// ---------------- K3: per-class mean features ----------------
__global__ void k_means(const float* __restrict__ f, const int* __restrict__ gt,
                        const unsigned* __restrict__ counts, float* __restrict__ means) {
    __shared__ int sgt[NOBJ];
    int l = blockIdx.x >> 3, k = blockIdx.x & 7;
    int tid = threadIdx.x;
    for (int i = tid; i < NOBJ; i += 256) sgt[i] = gt[i];
    __syncthreads();
    int d0 = tid, d1 = tid + 256;
    float a0 = 0.f, a1 = 0.f;
    const float* fl = f + (size_t)l * NOBJ * DIM;
    for (int n = 0; n < NOBJ; ++n) {
        if (sgt[n] == k) {
            const float* r = fl + (size_t)n * DIM;
            a0 += r[d0]; a1 += r[d1];
        }
    }
    unsigned c = counts[k];
    float inv = 1.f / (float)(c ? c : 1u);
    means[((size_t)(l * NC + k)) * DIM + d0] = a0 * inv;
    means[((size_t)(l * NC + k)) * DIM + d1] = a1 * inv;
}

// ---------------- K4: classification CE (one wave per row) ----------------
__global__ void k_clas(const float* __restrict__ f, const float* __restrict__ W,
                       const float* __restrict__ b, const int* __restrict__ gt,
                       float* __restrict__ cesum) {
    int wid  = (blockIdx.x * blockDim.x + threadIdx.x) >> 5;
    int lane = threadIdx.x & 31;
    if (wid >= NLVL * NOBJ) return;
    int l = wid >> 12, n = wid & 4095;
    const float* row = f + (size_t)wid * DIM;
    const float* Wl  = W + l * NC * DIM;
    float acc[NC] = {};
    for (int d = lane; d < DIM; d += 32) {
        float fv = row[d];
#pragma unroll
        for (int c = 0; c < NC; ++c) acc[c] += fv * Wl[c * DIM + d];
    }
#pragma unroll
    for (int c = 0; c < NC; ++c)
#pragma unroll
        for (int off = 16; off > 0; off >>= 1) acc[c] += __shfl_xor(acc[c], off, 32);
    if (lane == 0) {
        const float* bl = b + l * NC;
        float lg[NC], m = -1e30f;
#pragma unroll
        for (int c = 0; c < NC; ++c) { lg[c] = acc[c] + bl[c]; m = fmaxf(m, lg[c]); }
        float s = 0.f;
#pragma unroll
        for (int c = 0; c < NC; ++c) s += expf(lg[c] - m);
        float lse = m + logf(s);
        int g = gt[n];
        float lgg = 0.f;
#pragma unroll
        for (int c = 0; c < NC; ++c) lgg = (c == g) ? lg[c] : lgg;
        atomicAdd(&cesum[l * NC + g], lse - lgg);
    }
}

// ---------------- K5: finalize sim/dis/clas -> loss ----------------
__global__ void k_final(const unsigned* __restrict__ counts, const float* __restrict__ within,
                        const float* __restrict__ cesum, const float* __restrict__ means,
                        float* __restrict__ out) {
    __shared__ float G[36];
    __shared__ float lv[NLVL][3];
    int tid = threadIdx.x;
    for (int l = 0; l < NLVL; ++l) {
        if (tid < 36) {
            int i = 0, t = tid;
            while (t >= NC - i) { t -= NC - i; ++i; }
            int j = i + t;
            const float* mi = means + (size_t)(l * NC + i) * DIM;
            const float* mj = means + (size_t)(l * NC + j) * DIM;
            float s = 0.f;
            for (int d = 0; d < DIM; ++d) s += mi[d] * mj[d];
            G[tid] = s;
        }
        __syncthreads();
        if (tid == 0) {
            auto gidx = [](int i, int j) {
                int base = 0;
                for (int a = 0; a < i; ++a) base += NC - a;
                return base + (j - i);
            };
            float nrm[NC]; bool pres[NC]; int P = 0;
            for (int i = 0; i < NC; ++i) {
                nrm[i]  = sqrtf(fmaxf(G[gidx(i, i)], 0.f));
                pres[i] = counts[i] > 0u;
                P += pres[i] ? 1 : 0;
            }
            float dsum = 0.f;
            for (int i = 0; i < NC; ++i)
                for (int j = i + 1; j < NC; ++j)
                    if (pres[i] && pres[j]) {
                        float cs = G[gidx(i, j)] / fmaxf(nrm[i] * nrm[j], EPSF);
                        float tm = fmaxf(MARGINF - cs, 0.f);
                        dsum += tm * tm;
                    }
            int npc = (P * (P - 1)) / 2; if (npc < 1) npc = 1;
            float dis = (P >= 2) ? dsum / (float)npc : 0.f;
            float sim = 0.f, clas = 0.f;
            for (int k = 0; k < NC; ++k) {
                float c  = (float)counts[k];
                float np = c * (c - 1.f) * 0.5f;
                float ps = 0.5f * (within[l * NC + k] - c);   // diag S^2 == 1
                if (np > 0.f) sim += ps / np;
                if (counts[k] > 0u) clas += cesum[l * NC + k] / fmaxf(c, 1.f);
            }
            lv[l][0] = sim; lv[l][1] = dis; lv[l][2] = clas;
        }
        __syncthreads();
    }
    if (tid == 0) {
        float avg[3];
        for (int q = 0; q < 3; ++q) {
            float s = 0.f;
            for (int l = 0; l < NLVL; ++l) {
                float v = lv[l][q];
                if (v != v) v = TEMPF;          // NaN guard as in .average()
                s += v;
            }
            avg[q] = s / 3.f;
        }
        float loss = 0.002f * avg[0] + 0.005f * avg[1] + 0.005f * avg[2];
        out[0] = loss; out[1] = avg[0]; out[2] = avg[1]; out[3] = avg[2]; out[4] = loss;
    }
}

extern "C" void kernel_launch(void* const* d_in, const int* in_sizes, int n_in,
                              void* d_out, int out_size, void* d_ws, size_t ws_size,
                              hipStream_t stream) {
    (void)in_sizes; (void)n_in; (void)out_size;
    const float* f  = (const float*)d_in[0];   // [3,4096,512] f32
    const float* W  = (const float*)d_in[1];   // [3,8,512] f32
    const float* b  = (const float*)d_in[2];   // [3,8] f32
    const int*   gt = (const int*)d_in[3];     // [4096] i32

    char* ws = (char*)d_ws;
    unsigned* counts = (unsigned*)(ws + WS_COUNTS);
    float* within = (float*)(ws + WS_WITHIN);
    float* cesum  = (float*)(ws + WS_CESUM);
    float* rnorm  = (float*)(ws + WS_RNORM);
    float* means  = (float*)(ws + WS_MEANS);
    unsigned short* fb16 = (unsigned short*)(ws + WS_BF16);
    float* out    = (float*)d_out;

    const bool fast = ws_size >= WS_TOTAL;     // room for the bf16 mirror?

    k_init <<<1, 64, 0, stream>>>(counts, within, cesum);
    k_prep <<<(NLVL * NOBJ) / 8, 256, 0, stream>>>(f, gt, rnorm, counts,
                                                   fast ? fb16 : nullptr);
    if (fast)
        k_gram_async<<<NLVL * 64 * 64, 128, 0, stream>>>(fb16, gt, within);
    else
        k_gram_fb   <<<NLVL * 64 * 64, 128, 0, stream>>>(f, gt, rnorm, within);
    k_means<<<NLVL * NC, 256, 0, stream>>>(f, gt, counts, means);
    k_clas <<<(NLVL * NOBJ) / 8, 256, 0, stream>>>(f, W, b, gt, cesum);
    k_final<<<1, 64, 0, stream>>>(counts, within, cesum, means, out);
}
